// SafetyDistanceConstraint_54039278518798
// MI455X (gfx1250) — matrix operand order repaired and checked by reading, not compiled, not run on hardware
//
#include <hip/hip_runtime.h>
#include <math.h>

typedef _Float16 h16;
typedef __attribute__((ext_vector_type(16))) _Float16 v16h;
typedef __attribute__((ext_vector_type(8)))  _Float16 v8h;
typedef __attribute__((ext_vector_type(2)))  _Float16 v2h;
typedef __attribute__((ext_vector_type(8)))  float    v8f;

#define NB     64
#define NP     2016       // 64*63/2
#define HS     72         // h-tile row stride in halves (144 B, 16B-aligned)
#define WAVES  8
#define TILES  126        // NP / 16

#define ZERO8F {0.f, 0.f, 0.f, 0.f, 0.f, 0.f, 0.f, 0.f}

// p -> (i,j) for row-major upper triangle, N=64.  S(i) = i*(127-i)/2
__device__ __forceinline__ void pair_ij(int p, int& oi, int& oj) {
    int i = (int)((127.0f - sqrtf(16129.0f - 8.0f * (float)p)) * 0.5f);
    if (i < 0) i = 0;
    while ((i + 1) * (127 - (i + 1)) / 2 <= p) ++i;
    while (i > 0 && i * (127 - i) / 2 > p) --i;
    oi = i;
    oj = p - i * (127 - i) / 2 + i + 1;
}

// Packed-f16 epilogue: bias + relu + f16 convert, 2 rows at a time.
__device__ __forceinline__ void epilogue_frag(h16* __restrict__ hbase, const v8f& d,
                                              h16 bbh, int mo, int n) {
    union { v8f v; float e[8]; } ud; ud.v = d;
    const v2h bias2 = { bbh, bbh };
    const v2h zero2 = { (h16)0.f, (h16)0.f };
    #pragma unroll
    for (int r = 0; r < 8; r += 2) {
        v2h v = { (h16)ud.e[r], (h16)ud.e[r + 1] };   // v_cvt_pk_f16_f32
        v = v + bias2;                                 // v_pk_add_f16
        v = __builtin_elementwise_max(v, zero2);       // v_pk_max_num_f16
        hbase[(mo + r) * HS + n]     = v.x;
        hbase[(mo + r + 1) * HS + n] = v.y;
    }
}

// ---------------------------------------------------------------------------
// Kernel 1: margin MLP (6->64->64->1), violations + pair_active flags
// ---------------------------------------------------------------------------
__global__ __launch_bounds__(256)
void sdc_violations(const float* __restrict__ pos, const float* __restrict__ vel,
                    const float* __restrict__ w1, const float* __restrict__ b1,
                    const float* __restrict__ w2, const float* __restrict__ b2,
                    const float* __restrict__ w3, const float* __restrict__ b3,
                    float* __restrict__ out_viol, int* __restrict__ active) {
    __shared__ float s_pos[NB * 3], s_vel[NB * 3];
    __shared__ float s_w1[6 * 64], s_w3[64];
    __shared__ h16   s_b1h[64], s_b2h[64];
    __shared__ float s_b3s;
    __shared__ h16   s_w2[64 * 64];
    __shared__ h16   s_h[WAVES][16 * HS];
    __shared__ unsigned short s_ij[NP];

    const int tid  = threadIdx.x;
    const int b    = blockIdx.x;
    const int lane = tid & 31;
    const int w    = tid >> 5;
    const int m    = lane & 15;
    const int hi   = lane >> 4;

    if (tid < NB * 3) { s_pos[tid] = pos[(size_t)b * NB * 3 + tid];
                        s_vel[tid] = vel[(size_t)b * NB * 3 + tid]; }
    for (int k = tid; k < 6 * 64; k += 256) s_w1[k] = w1[k];
    if (tid < 64) { s_b1h[tid] = (h16)b1[tid]; s_b2h[tid] = (h16)b2[tid];
                    s_w3[tid] = w3[tid]; }
    if (tid == 0) s_b3s = b3[0];
    for (int k = tid; k < 64 * 64; k += 256) s_w2[k] = (h16)w2[k];
    for (int p0 = tid; p0 < NP; p0 += 256) {
        int pi, pj; pair_ij(p0, pi, pj);
        s_ij[p0] = (unsigned short)(pi | (pj << 8));
    }
    __syncthreads();

    // Layer-1 B fragments: W1 zero-padded to 32x64
    v16h bf1[4];
    {
        const int n = m, kb = hi * 16;
        for (int jn = 0; jn < 4; ++jn) {
            union { v16h v; h16 e[16]; } u;
            #pragma unroll
            for (int tt = 0; tt < 16; ++tt) {
                const int K = kb + tt;
                u.e[tt] = (K < 6) ? (h16)s_w1[K * 64 + jn * 16 + n] : (h16)0.f;
            }
            bf1[jn] = u.v;
        }
    }
    // Layer-2 B fragments: W2 64x64 -> 2 K-chunks x 4 N-chunks of 32x16
    v16h bf2[2][4];
    {
        const int n = m, kb = hi * 16;
        for (int c = 0; c < 2; ++c)
            for (int jn = 0; jn < 4; ++jn) {
                union { v16h v; h16 e[16]; } u;
                #pragma unroll
                for (int tt = 0; tt < 16; ++tt)
                    u.e[tt] = s_w2[(c * 32 + kb + tt) * 64 + jn * 16 + n];
                bf2[c][jn] = u.v;
            }
    }

    for (int t = w; t < TILES; t += WAVES) {
        const int p  = t * 16 + m;
        const int ij = s_ij[p];
        const int i  = ij & 0xff;
        const int j  = ij >> 8;
        const float px = s_pos[j*3+0] - s_pos[i*3+0];
        const float py = s_pos[j*3+1] - s_pos[i*3+1];
        const float pz = s_pos[j*3+2] - s_pos[i*3+2];
        const float vx = s_vel[j*3+0] - s_vel[i*3+0];
        const float vy = s_vel[j*3+1] - s_vel[i*3+1];
        const float vz = s_vel[j*3+2] - s_vel[i*3+2];
        const float dist = sqrtf(px*px + py*py + pz*pz);

        // Layer-1 A fragment built directly in registers (K=0..5 on lanes 0-15)
        const bool lo = (hi == 0);
        union { v16h v; h16 e[16]; } ux;
        #pragma unroll
        for (int tt = 0; tt < 16; ++tt) ux.e[tt] = (h16)0.f;
        ux.e[0] = lo ? (h16)px : (h16)0.f;
        ux.e[1] = lo ? (h16)py : (h16)0.f;
        ux.e[2] = lo ? (h16)pz : (h16)0.f;
        ux.e[3] = lo ? (h16)vx : (h16)0.f;
        ux.e[4] = lo ? (h16)vy : (h16)0.f;
        ux.e[5] = lo ? (h16)vz : (h16)0.f;

        // Layer 1: issue all 4 WMMAs, then epilogues (hazard distance covered)
        v8f d1[4];
        #pragma unroll
        for (int jn = 0; jn < 4; ++jn) {
            const v8f z = ZERO8F;
            d1[jn] = __builtin_amdgcn_wmma_f32_16x16x32_f16(false, ux.v, false, bf1[jn],
                                                            (short)0, z, false, false);
        }
        const int mo = hi * 8;
        #pragma unroll
        for (int jn = 0; jn < 4; ++jn)
            epilogue_frag(s_h[w], d1[jn], s_b1h[jn * 16 + m], mo, jn * 16 + m);

        // A-fragments for layer 2 from LDS (16x32 f16 WMMA layout)
        h16* hrow = &s_h[w][m * HS];
        v16h a[2];
        #pragma unroll
        for (int c = 0; c < 2; ++c) {
            union { v16h v; v8h h8[2]; } ua;
            ua.h8[0] = *(const v8h*)&hrow[c * 32 + hi * 16];
            ua.h8[1] = *(const v8h*)&hrow[c * 32 + hi * 16 + 8];
            a[c] = ua.v;
        }

        // Layer 2: 8 WMMAs back-to-back (K-chunk 0 then accumulate K-chunk 1)
        v8f d2[4];
        #pragma unroll
        for (int jn = 0; jn < 4; ++jn) {
            const v8f z = ZERO8F;
            d2[jn] = __builtin_amdgcn_wmma_f32_16x16x32_f16(false, a[0], false, bf2[0][jn],
                                                            (short)0, z, false, false);
        }
        #pragma unroll
        for (int jn = 0; jn < 4; ++jn)
            d2[jn] = __builtin_amdgcn_wmma_f32_16x16x32_f16(false, a[1], false, bf2[1][jn],
                                                            (short)0, d2[jn], false, false);
        #pragma unroll
        for (int jn = 0; jn < 4; ++jn)
            epilogue_frag(s_h[w], d2[jn], s_b2h[jn * 16 + m], mo, jn * 16 + m);

        // Layer 3 (lanes 0..15): sigmoid head -> violation
        if (hi == 0) {
            float acc = s_b3s;
            const h16* hr = &s_h[w][m * HS];
            #pragma unroll
            for (int k8 = 0; k8 < 64; k8 += 8) {
                union { v8h v; h16 e[8]; } uh;
                uh.v = *(const v8h*)&hr[k8];
                #pragma unroll
                for (int q = 0; q < 8; ++q)
                    acc = fmaf((float)uh.e[q], s_w3[k8 + q], acc);
            }
            const float sg   = 1.0f / (1.0f + __expf(-acc));
            const float viol = 5.0f * (1.0f + sg) - dist;
            out_viol[(size_t)b * NP + p] = viol;
            if (viol > 0.0f) atomicOr(&active[p], 1);
        }
    }
}

// ---------------------------------------------------------------------------
// Kernel 2: repulsion MLP (7->64->64->3), masked scatter-add of forces
// ---------------------------------------------------------------------------
__global__ __launch_bounds__(256)
void sdc_forces(const float* __restrict__ pos, const float* __restrict__ vel,
                const float* __restrict__ w1, const float* __restrict__ b1,
                const float* __restrict__ w2, const float* __restrict__ b2,
                const float* __restrict__ w3, const float* __restrict__ b3,
                const int* __restrict__ active, float* __restrict__ out_forces) {
    __shared__ float s_pos[NB * 3], s_vel[NB * 3];
    __shared__ float s_w1[7 * 64], s_w3[64 * 3], s_b3[3];
    __shared__ h16   s_b1h[64], s_b2h[64];
    __shared__ h16   s_w2[64 * 64];
    __shared__ h16   s_h[WAVES][16 * HS];
    __shared__ float s_f[NB * 3];
    __shared__ unsigned short s_ij[NP];

    const int tid  = threadIdx.x;
    const int b    = blockIdx.x;
    const int lane = tid & 31;
    const int w    = tid >> 5;
    const int m    = lane & 15;
    const int hi   = lane >> 4;

    if (tid < NB * 3) { s_pos[tid] = pos[(size_t)b * NB * 3 + tid];
                        s_vel[tid] = vel[(size_t)b * NB * 3 + tid];
                        s_f[tid]   = 0.0f; }
    for (int k = tid; k < 7 * 64; k += 256) s_w1[k] = w1[k];
    if (tid < 64) { s_b1h[tid] = (h16)b1[tid]; s_b2h[tid] = (h16)b2[tid]; }
    for (int k = tid; k < 64 * 3; k += 256) s_w3[k] = w3[k];
    if (tid < 3) s_b3[tid] = b3[tid];
    for (int k = tid; k < 64 * 64; k += 256) s_w2[k] = (h16)w2[k];
    for (int p0 = tid; p0 < NP; p0 += 256) {
        int pi, pj; pair_ij(p0, pi, pj);
        s_ij[p0] = (unsigned short)(pi | (pj << 8));
    }
    __syncthreads();

    v16h bf1[4];
    {
        const int n = m, kb = hi * 16;
        for (int jn = 0; jn < 4; ++jn) {
            union { v16h v; h16 e[16]; } u;
            #pragma unroll
            for (int tt = 0; tt < 16; ++tt) {
                const int K = kb + tt;
                u.e[tt] = (K < 7) ? (h16)s_w1[K * 64 + jn * 16 + n] : (h16)0.f;
            }
            bf1[jn] = u.v;
        }
    }
    v16h bf2[2][4];
    {
        const int n = m, kb = hi * 16;
        for (int c = 0; c < 2; ++c)
            for (int jn = 0; jn < 4; ++jn) {
                union { v16h v; h16 e[16]; } u;
                #pragma unroll
                for (int tt = 0; tt < 16; ++tt)
                    u.e[tt] = s_w2[(c * 32 + kb + tt) * 64 + jn * 16 + n];
                bf2[c][jn] = u.v;
            }
    }

    for (int t = w; t < TILES; t += WAVES) {
        const int p  = t * 16 + m;
        const int ij = s_ij[p];
        const int i  = ij & 0xff;
        const int j  = ij >> 8;
        const float px = s_pos[j*3+0] - s_pos[i*3+0];
        const float py = s_pos[j*3+1] - s_pos[i*3+1];
        const float pz = s_pos[j*3+2] - s_pos[i*3+2];
        const float vx = s_vel[j*3+0] - s_vel[i*3+0];
        const float vy = s_vel[j*3+1] - s_vel[i*3+1];
        const float vz = s_vel[j*3+2] - s_vel[i*3+2];
        const float dist = sqrtf(px*px + py*py + pz*pz);

        const bool lo = (hi == 0);
        union { v16h v; h16 e[16]; } ux;
        #pragma unroll
        for (int tt = 0; tt < 16; ++tt) ux.e[tt] = (h16)0.f;
        ux.e[0] = lo ? (h16)px   : (h16)0.f;
        ux.e[1] = lo ? (h16)py   : (h16)0.f;
        ux.e[2] = lo ? (h16)pz   : (h16)0.f;
        ux.e[3] = lo ? (h16)vx   : (h16)0.f;
        ux.e[4] = lo ? (h16)vy   : (h16)0.f;
        ux.e[5] = lo ? (h16)vz   : (h16)0.f;
        ux.e[6] = lo ? (h16)dist : (h16)0.f;

        v8f d1[4];
        #pragma unroll
        for (int jn = 0; jn < 4; ++jn) {
            const v8f z = ZERO8F;
            d1[jn] = __builtin_amdgcn_wmma_f32_16x16x32_f16(false, ux.v, false, bf1[jn],
                                                            (short)0, z, false, false);
        }
        const int mo = hi * 8;
        #pragma unroll
        for (int jn = 0; jn < 4; ++jn)
            epilogue_frag(s_h[w], d1[jn], s_b1h[jn * 16 + m], mo, jn * 16 + m);

        h16* hrow = &s_h[w][m * HS];
        v16h a[2];
        #pragma unroll
        for (int c = 0; c < 2; ++c) {
            union { v16h v; v8h h8[2]; } ua;
            ua.h8[0] = *(const v8h*)&hrow[c * 32 + hi * 16];
            ua.h8[1] = *(const v8h*)&hrow[c * 32 + hi * 16 + 8];
            a[c] = ua.v;
        }

        v8f d2[4];
        #pragma unroll
        for (int jn = 0; jn < 4; ++jn) {
            const v8f z = ZERO8F;
            d2[jn] = __builtin_amdgcn_wmma_f32_16x16x32_f16(false, a[0], false, bf2[0][jn],
                                                            (short)0, z, false, false);
        }
        #pragma unroll
        for (int jn = 0; jn < 4; ++jn)
            d2[jn] = __builtin_amdgcn_wmma_f32_16x16x32_f16(false, a[1], false, bf2[1][jn],
                                                            (short)0, d2[jn], false, false);
        #pragma unroll
        for (int jn = 0; jn < 4; ++jn)
            epilogue_frag(s_h[w], d2[jn], s_b2h[jn * 16 + m], mo, jn * 16 + m);

        if (hi == 0) {
            float r0 = s_b3[0], r1 = s_b3[1], r2 = s_b3[2];
            const h16* hr = &s_h[w][m * HS];
            #pragma unroll
            for (int k8 = 0; k8 < 64; k8 += 8) {
                union { v8h v; h16 e[8]; } uh;
                uh.v = *(const v8h*)&hr[k8];
                #pragma unroll
                for (int q = 0; q < 8; ++q) {
                    const float hv = (float)uh.e[q];
                    r0 = fmaf(hv, s_w3[(k8 + q) * 3 + 0], r0);
                    r1 = fmaf(hv, s_w3[(k8 + q) * 3 + 1], r1);
                    r2 = fmaf(hv, s_w3[(k8 + q) * 3 + 2], r2);
                }
            }
            if (active[p]) {
                atomicAdd(&s_f[i * 3 + 0], -r0);
                atomicAdd(&s_f[i * 3 + 1], -r1);
                atomicAdd(&s_f[i * 3 + 2], -r2);
                atomicAdd(&s_f[j * 3 + 0],  r0);
                atomicAdd(&s_f[j * 3 + 1],  r1);
                atomicAdd(&s_f[j * 3 + 2],  r2);
            }
        }
    }

    __syncthreads();
    if (tid < NB * 3) out_forces[(size_t)b * NB * 3 + tid] = s_f[tid];
}

// ---------------------------------------------------------------------------
extern "C" void kernel_launch(void* const* d_in, const int* in_sizes, int n_in,
                              void* d_out, int out_size, void* d_ws, size_t ws_size,
                              hipStream_t stream) {
    const float* pos   = (const float*)d_in[0];
    const float* vel   = (const float*)d_in[1];
    const float* sm_w1 = (const float*)d_in[2];
    const float* sm_b1 = (const float*)d_in[3];
    const float* sm_w2 = (const float*)d_in[4];
    const float* sm_b2 = (const float*)d_in[5];
    const float* sm_w3 = (const float*)d_in[6];
    const float* sm_b3 = (const float*)d_in[7];
    const float* rp_w1 = (const float*)d_in[8];
    const float* rp_b1 = (const float*)d_in[9];
    const float* rp_w2 = (const float*)d_in[10];
    const float* rp_b2 = (const float*)d_in[11];
    const float* rp_w3 = (const float*)d_in[12];
    const float* rp_b3 = (const float*)d_in[13];

    const int B = in_sizes[0] / (NB * 3);
    float* out        = (float*)d_out;
    float* out_forces = out;
    float* out_viol   = out + (size_t)B * NB * 3;
    int*   active     = (int*)d_ws;

    hipMemsetAsync(active, 0, NP * sizeof(int), stream);

    sdc_violations<<<B, 256, 0, stream>>>(pos, vel, sm_w1, sm_b1, sm_w2, sm_b2,
                                          sm_w3, sm_b3, out_viol, active);
    sdc_forces<<<B, 256, 0, stream>>>(pos, vel, rp_w1, rp_b1, rp_w2, rp_b2,
                                      rp_w3, rp_b3, active, out_forces);
}
